// MemoryGraph_68985764708404
// MI455X (gfx1250) — matrix operand order repaired and checked by reading, not compiled, not run on hardware
//
#include <hip/hip_runtime.h>
#include <hip/hip_bf16.h>

// ---------------- problem constants (from setup_inputs) ----------------
#define BSZ 4      // batch
#define NN  4096   // neurons
#define DD  32     // state dim
#define KK  64     // connections per neuron
#define HH  256    // H_state == H_msg
#define HM  64     // H_mod
#define MOD_IN  129
#define MOD_OUT 97
#define TT  16     // T (fixed by setup_inputs; device scalar d_in[21] not host-readable under graph capture)
#define CMEM 64

typedef __attribute__((ext_vector_type(16))) _Float16 v16h;
typedef __attribute__((ext_vector_type(8)))  _Float16 v8h;
typedef __attribute__((ext_vector_type(8)))  float    v8f;

// Branch-free tanh: tanh(x) = sign(x) * (1 - 2/(exp(2|x|)+1)).
// v_exp_f32 + v_rcp_f32, no exec-mask divergence (libm tanhf emits branchy code
// that interleaves s_and_saveexec with the WMMA stream).
static __device__ __forceinline__ float fast_tanh(float x) {
    float ax = fabsf(x);
    float e  = __expf(2.0f * ax);
    float t  = 1.0f - 2.0f * __builtin_amdgcn_rcpf(e + 1.0f);
    return copysignf(t, x);
}
static __device__ __forceinline__ float sigm(float x) {
    return __builtin_amdgcn_rcpf(1.0f + __expf(-x));
}

// A-matrix fragment for V_WMMA_F32_16X16X32_F16 (16x32, f16), contiguous k order (not the ISA table order):
// lane L holds row M = L&15; halves 0..15 = K = (L>>4)*16 .. +15 (matches load_b_frag)
static __device__ __forceinline__ v16h load_a_frag(const _Float16* base, int rowstride, int kofs, int lane) {
    int row = lane & 15;
    int k0  = kofs + ((lane >> 4) << 4);
    v8h lo = *(const v8h*)(base + row * rowstride + k0);
    v8h hi = *(const v8h*)(base + row * rowstride + k0 + 8);
    v16h a;
#pragma unroll
    for (int i = 0; i < 8; ++i) { a[i] = lo[i]; a[8 + i] = hi[i]; }
    return a;
}

// B-matrix fragment (32x16, f16): lane L holds column N = L&15, halves = K = (L>>4)*16 .. +15.
// Weights stored column-contiguous (column n = `colstride` consecutive halves) -> one 32B load.
static __device__ __forceinline__ v16h load_b_frag(const _Float16* colbase, int colstride, int ntile, int kofs, int lane) {
    int col = (ntile << 4) + (lane & 15);
    int k0  = kofs + ((lane >> 4) << 4);
    return *(const v16h*)(colbase + col * colstride + k0);
}

static __device__ __forceinline__ v8f wmma16(v16h a, v16h b, v8f c) {
    return __builtin_amdgcn_wmma_f32_16x16x32_f16(false, a, false, b, (short)0, c, false, false);
}

// ---------------- init: copy state, zero accumulators ----------------
__global__ __launch_bounds__(256) void k_init(const float* __restrict__ h0, const float* __restrict__ pm0,
                                              float* __restrict__ h_g, float* __restrict__ msg0,
                                              float* __restrict__ heb, float* __restrict__ mem_out) {
    int idx = blockIdx.x * 256 + threadIdx.x;           // grid covers BSZ*NN*KK = 1048576
    if (idx < BSZ * NN * DD) { h_g[idx] = h0[idx]; msg0[idx] = pm0[idx]; }
    if (idx < BSZ * NN * KK) heb[idx] = 0.0f;
    if (idx < BSZ * TT * CMEM * DD) mem_out[idx] = 0.0f;
}

// ---------------- prep: convert recurrent weights to f16 B-layouts ----------------
// winT[h*32+d]  = state_w1[h][d]      (B for D->H state matmul, column h contiguous)
// sw2h[d*256+h] = state_w2[d][h]      (B for H->D state matmul)
// mwhh[h*32+d]  = msg_w1[h][d]        (B for D->H msg matmul)
// mw2h[d*256+h] = msg_w2[d][h]        (B for H->D msg matmul)
__global__ __launch_bounds__(256) void k_prep(const float* __restrict__ sw1, const float* __restrict__ sw2,
                                              const float* __restrict__ msw1, const float* __restrict__ msw2,
                                              _Float16* __restrict__ winT, _Float16* __restrict__ sw2h,
                                              _Float16* __restrict__ mwhh, _Float16* __restrict__ mw2h) {
    int idx = blockIdx.x * 256 + threadIdx.x;           // grid covers 8192
    if (idx < HH * DD) {
        int h = idx >> 5, d = idx & 31;
        winT[idx] = (_Float16)sw1[h * 97 + d];
        mwhh[idx] = (_Float16)msw1[h * 96 + d];
        sw2h[idx] = (_Float16)sw2[idx];
        mw2h[idx] = (_Float16)msw2[idx];
    }
}

// ---------------- modulator MLP + bias folding (one block per neuron) ----------------
__global__ __launch_bounds__(256) void k_mod(const float* __restrict__ h0, const float* __restrict__ dlog,
                                             const float* __restrict__ prim0, const float* __restrict__ heb0,
                                             const float* __restrict__ mw1, const float* __restrict__ mb1,
                                             const float* __restrict__ mw2, const float* __restrict__ mb2,
                                             const float* __restrict__ sw1, const float* __restrict__ sb1,
                                             const float* __restrict__ msw1, const float* __restrict__ msb1,
                                             const float* __restrict__ nid,
                                             float* __restrict__ w_sig, float* __restrict__ decay,
                                             float* __restrict__ sbias, float* __restrict__ mbias) {
    int n = blockIdx.x;
    int tid = threadIdx.x;
    __shared__ float lx[BSZ][132];     // mod_input (padded)
    __shared__ float lh[BSZ][HM];      // hidden
    __shared__ float lprim[BSZ][DD];
    __shared__ float ldec[BSZ];

    for (int idx = tid; idx < BSZ * MOD_IN; idx += 256) {
        int bb = idx / MOD_IN, i = idx % MOD_IN;
        float v;
        if (i < KK)            v = heb0[(bb * NN + n) * KK + i];
        else if (i < KK + DD)  v = h0[(bb * NN + n) * DD + (i - KK)];
        else if (i == KK + DD) v = dlog[bb * NN + n];
        else                   v = prim0[(bb * NN + n) * DD + (i - KK - DD - 1)];
        lx[bb][i] = v;
    }
    __syncthreads();
    {   // hidden = tanh(W1 x + b1): 256 threads = (b in 0..3) x (h in 0..63)
        int bb = tid >> 6, hh = tid & 63;
        const float* wr = mw1 + (size_t)(n * HM + hh) * MOD_IN;
        float acc = mb1[n * HM + hh];
        for (int i = 0; i < MOD_IN; ++i) acc += lx[bb][i] * wr[i];
        lh[bb][hh] = fast_tanh(acc);
    }
    __syncthreads();
    for (int idx = tid; idx < BSZ * MOD_OUT; idx += 256) {
        int bb = idx / MOD_OUT, o = idx % MOD_OUT;
        float acc = mb2[n * MOD_OUT + o];
        for (int h = 0; h < HM; ++h) acc += lh[bb][h] * mw2[(size_t)(n * HM + h) * MOD_OUT + o];
        if (o < KK)            w_sig[(bb * NN + n) * KK + o] = sigm(acc);
        else if (o == KK)      { float dv = sigm(acc); decay[bb * NN + n] = dv; ldec[bb] = dv; }
        else                   lprim[bb][o - KK - 1] = acc;
    }
    __syncthreads();
    // state_bias = prim@w_prim_s^T + id@w_id_s^T + decay*w_decay + state_b1 ; msg_bias analogous
    for (int idx = tid; idx < BSZ * HH; idx += 256) {
        int bb = idx / HH, h = idx % HH;
        const float* s1 = sw1 + h * 97;
        const float* m1 = msw1 + h * 96;
        float sa = sb1[h] + ldec[bb] * s1[96];
        float ma = msb1[h];
#pragma unroll
        for (int d = 0; d < DD; ++d) {
            float p = lprim[bb][d], id = nid[n * DD + d];
            sa += p * s1[32 + d] + id * s1[64 + d];
            ma += p * m1[32 + d] + id * m1[64 + d];
        }
        sbias[(bb * NN + n) * HH + h] = sa;
        mbias[(bb * NN + n) * HH + h] = ma;
    }
}

// ---------------- one recurrent step: 4 waves/block, each wave owns 16 neurons ----------------
__global__ __launch_bounds__(128) void k_step(int t, int do_heb,
                                              const float* __restrict__ prev_msg, float* __restrict__ msg_out,
                                              const float* __restrict__ mag_prev, float* __restrict__ mag_out,
                                              float* __restrict__ h_g,
                                              const int* __restrict__ conn,
                                              const float* __restrict__ w_sig,
                                              const float* __restrict__ sbias, const float* __restrict__ mbias,
                                              const float* __restrict__ decay,
                                              const float* __restrict__ dbw, const float* __restrict__ dgw,
                                              const _Float16* __restrict__ winT, const _Float16* __restrict__ sw2h,
                                              const _Float16* __restrict__ mwhh, const _Float16* __restrict__ mw2h,
                                              const float* __restrict__ state_b2, const float* __restrict__ msg_b2,
                                              float* __restrict__ mem_out, float* __restrict__ heb_sum) {
    __shared__ _Float16 ldsA[4][16 * 32];    // recv, later h_new (f16 A tiles)
    __shared__ _Float16 ldsH[4][16 * 256];   // hid, later mh (f16 A tiles)
    __shared__ float    ldsM[4][16 * 32];    // msg (f32, for reductions)

    const int tid = threadIdx.x, wave = tid >> 5, lane = tid & 31;
    const int b  = blockIdx.y;
    const int n0 = (blockIdx.x * 4 + wave) * 16;
    const int d  = lane;                      // 0..31 == DD

    // Keep the recurrent weight matrices (4 x 16KB) hot in cache: one 128B line per thread.
    {
        size_t lo = (size_t)tid * 128;
        __builtin_prefetch((const char*)winT + lo, 0, 0);
        __builtin_prefetch((const char*)sw2h + lo, 0, 0);
        __builtin_prefetch((const char*)mwhh + lo, 0, 0);
        __builtin_prefetch((const char*)mw2h + lo, 0, 0);
    }

    // ---- dendritic gather/tree: recv[r][d] ----
    for (int r = 0; r < 16; ++r) {
        int n = n0 + r;
        const int*   cr  = conn + n * KK;
        const float* wsr = w_sig + (size_t)(b * NN + n) * KK;
        const float* dbr = dbw + (size_t)n * (8 * 8 * DD);
        const float* dgr = dgw + (size_t)n * (2 * 4 * DD);
        float recv = 0.0f;
#pragma unroll
        for (int g = 0; g < 2; ++g) {
            float gacc = 0.0f;
#pragma unroll
            for (int p = 0; p < 4; ++p) {
                int j = g * 4 + p;
                float bacc = 0.0f;
#pragma unroll
                for (int s = 0; s < 8; ++s) {
                    int k = j * 8 + s;
                    int src = cr[k];
                    bacc += prev_msg[(size_t)(b * NN + src) * DD + d] * wsr[k] * dbr[(j * 8 + s) * DD + d];
                }
                gacc += fast_tanh(bacc) * dgr[(g * 4 + p) * DD + d];
            }
            recv += fast_tanh(gacc);
        }
        ldsA[wave][r * 32 + d] = (_Float16)(0.5f * recv);
    }

    // ---- hebbian accumulation for previous step's messages ----
    if (do_heb) {
        for (int idx = lane; idx < 16 * KK; idx += 32) {
            int r = idx >> 6, k = idx & 63;
            int n = n0 + r;
            int src = conn[n * KK + k];
            size_t o = (size_t)(b * NN + n) * KK + k;
            heb_sum[o] += mag_prev[b * NN + src] * w_sig[o];
        }
    }
    __syncthreads();

    // ---- matmul1: hid = tanh(recv @ w_in^T + state_bias)  (16x32)*(32x256) ----
    {
        v16h aA = load_a_frag(ldsA[wave], 32, 0, lane);
#pragma unroll
        for (int nt = 0; nt < 16; ++nt) {
            v16h bF = load_b_frag(winT, 32, nt, 0, lane);
            v8f c = {};
            c = wmma16(aA, bF, c);
            int col = nt * 16 + (lane & 15);
#pragma unroll
            for (int v = 0; v < 8; ++v) {
                int row = v + ((lane >> 4) << 3);
                float val = c[v] + sbias[(size_t)(b * NN + n0 + row) * HH + col];
                ldsH[wave][row * 256 + col] = (_Float16)fast_tanh(val);
            }
        }
    }
    __syncthreads();

    // ---- matmul2: update = tanh(hid @ state_w2^T + b2); h_new = d*h + (1-d)*update ----
#pragma unroll
    for (int dt = 0; dt < 2; ++dt) {
        v8f c = {};
#pragma unroll
        for (int kc = 0; kc < 8; ++kc) {
            v16h a  = load_a_frag(ldsH[wave], 256, kc * 32, lane);
            v16h bF = load_b_frag(sw2h, 256, dt, kc * 32, lane);
            c = wmma16(a, bF, c);
        }
        int dcol = dt * 16 + (lane & 15);
#pragma unroll
        for (int v = 0; v < 8; ++v) {
            int row = v + ((lane >> 4) << 3);
            int n = n0 + row;
            float upd  = fast_tanh(c[v] + state_b2[dcol]);
            float dec  = decay[b * NN + n];
            size_t o   = (size_t)(b * NN + n) * DD + dcol;
            float hn   = dec * h_g[o] + (1.0f - dec) * upd;
            h_g[o] = hn;
            ldsA[wave][row * 32 + dcol] = (_Float16)hn;
        }
    }
    __syncthreads();

    // ---- matmul3: mh = tanh(h_new @ mw_h^T + msg_bias) ----
    {
        v16h aH = load_a_frag(ldsA[wave], 32, 0, lane);
#pragma unroll
        for (int nt = 0; nt < 16; ++nt) {
            v16h bF = load_b_frag(mwhh, 32, nt, 0, lane);
            v8f c = {};
            c = wmma16(aH, bF, c);
            int col = nt * 16 + (lane & 15);
#pragma unroll
            for (int v = 0; v < 8; ++v) {
                int row = v + ((lane >> 4) << 3);
                float val = c[v] + mbias[(size_t)(b * NN + n0 + row) * HH + col];
                ldsH[wave][row * 256 + col] = (_Float16)fast_tanh(val);
            }
        }
    }
    __syncthreads();

    // ---- matmul4: msg = tanh(mh @ msg_w2^T + msg_b2) ----
#pragma unroll
    for (int dt = 0; dt < 2; ++dt) {
        v8f c = {};
#pragma unroll
        for (int kc = 0; kc < 8; ++kc) {
            v16h a  = load_a_frag(ldsH[wave], 256, kc * 32, lane);
            v16h bF = load_b_frag(mw2h, 256, dt, kc * 32, lane);
            c = wmma16(a, bF, c);
        }
        int dcol = dt * 16 + (lane & 15);
#pragma unroll
        for (int v = 0; v < 8; ++v) {
            int row = v + ((lane >> 4) << 3);
            int n = n0 + row;
            float m = fast_tanh(c[v] + msg_b2[dcol]);
            msg_out[(size_t)(b * NN + n) * DD + dcol] = m;
            ldsM[wave][row * 32 + dcol] = m;
        }
    }
    __syncthreads();

    // ---- reductions: msg magnitude per neuron; mem slice partial sums ----
    if (lane < 16) {
        int row = lane;
        float s = 0.0f;
#pragma unroll
        for (int dd = 0; dd < 32; ++dd) s += fabsf(ldsM[wave][row * 32 + dd]);
        mag_out[b * NN + n0 + row] = s * (1.0f / 32.0f);
    }
    {
        float s = 0.0f;
#pragma unroll
        for (int r = 0; r < 16; ++r) s += ldsM[wave][r * 32 + lane];
        int cs = n0 >> 6;   // all 16 neurons of this wave share one C_MEM slice
        atomicAdd(&mem_out[(((size_t)b * TT + t) * CMEM + cs) * DD + lane], s * 0.125f);  // 64^-0.5
    }
}

// ---------------- finalize: last hebbian contribution, /T, copy h ----------------
__global__ __launch_bounds__(256) void k_fin(const float* __restrict__ heb_sum, const float* __restrict__ mag_last,
                                             const float* __restrict__ w_sig, const int* __restrict__ conn,
                                             const float* __restrict__ h_g,
                                             float* __restrict__ out_h, float* __restrict__ out_heb) {
    int idx = blockIdx.x * 256 + threadIdx.x;           // grid covers BSZ*NN*KK
    if (idx < BSZ * NN * KK) {
        int bb  = idx / (NN * KK);
        int rem = idx - bb * NN * KK;
        int n   = rem / KK, k = rem % KK;
        int src = conn[n * KK + k];
        out_heb[idx] = (heb_sum[idx] + mag_last[bb * NN + src] * w_sig[idx]) * (1.0f / (float)TT);
    }
    if (idx < BSZ * NN * DD) out_h[idx] = h_g[idx];
}

extern "C" void kernel_launch(void* const* d_in, const int* in_sizes, int n_in,
                              void* d_out, int out_size, void* d_ws, size_t ws_size,
                              hipStream_t stream) {
    (void)in_sizes; (void)n_in; (void)out_size; (void)ws_size;
    const float* h0    = (const float*)d_in[0];
    const float* pm0   = (const float*)d_in[1];
    const float* dlog  = (const float*)d_in[2];
    const float* prim0 = (const float*)d_in[3];
    const float* heb0  = (const float*)d_in[4];
    const float* mw1   = (const float*)d_in[5];
    const float* mb1   = (const float*)d_in[6];
    const float* mw2   = (const float*)d_in[7];
    const float* mb2   = (const float*)d_in[8];
    const float* sw1   = (const float*)d_in[9];
    const float* sb1   = (const float*)d_in[10];
    const float* sw2   = (const float*)d_in[11];
    const float* sb2   = (const float*)d_in[12];
    const float* msw1  = (const float*)d_in[13];
    const float* msb1  = (const float*)d_in[14];
    const float* msw2  = (const float*)d_in[15];
    const float* msb2  = (const float*)d_in[16];
    const float* nid   = (const float*)d_in[17];
    const float* dbw   = (const float*)d_in[18];
    const float* dgw   = (const float*)d_in[19];
    const int*   conn  = (const int*)d_in[20];
    // d_in[21] = T (device scalar); TT=16 fixed by setup_inputs.

    // ---- workspace carve-up (floats) ----
    float* ws = (float*)d_ws;
    size_t off = 0;
    float* w_sig = ws + off; off += (size_t)BSZ * NN * KK;     // 1,048,576
    float* sbias = ws + off; off += (size_t)BSZ * NN * HH;     // 4,194,304
    float* mbias = ws + off; off += (size_t)BSZ * NN * HH;     // 4,194,304
    float* decay = ws + off; off += (size_t)BSZ * NN;          // 16,384
    float* h_g   = ws + off; off += (size_t)BSZ * NN * DD;     // 524,288
    float* msgb0 = ws + off; off += (size_t)BSZ * NN * DD;
    float* msgb1 = ws + off; off += (size_t)BSZ * NN * DD;
    float* heb   = ws + off; off += (size_t)BSZ * NN * KK;
    float* mag0  = ws + off; off += (size_t)BSZ * NN;
    float* mag1  = ws + off; off += (size_t)BSZ * NN;
    _Float16* f16b = (_Float16*)(ws + off);                     // 32B-aligned
    _Float16* winT = f16b;
    _Float16* sw2h = f16b + 8192;
    _Float16* mwhh = f16b + 16384;
    _Float16* mw2h = f16b + 24576;
    float* msgb[2] = {msgb0, msgb1};
    float* magb[2] = {mag0, mag1};

    // ---- output carve-up: mem(4,16,64,32) | h_fin(4,4096,32) | hebbian(4,4096,64) ----
    float* out_mem = (float*)d_out;
    float* out_h   = out_mem + (size_t)BSZ * TT * CMEM * DD;
    float* out_heb = out_h + (size_t)BSZ * NN * DD;

    k_init<<<4096, 256, 0, stream>>>(h0, pm0, h_g, msgb0, heb, out_mem);
    k_prep<<<32, 256, 0, stream>>>(sw1, sw2, msw1, msw2, winT, sw2h, mwhh, mw2h);
    k_mod<<<NN, 256, 0, stream>>>(h0, dlog, prim0, heb0, mw1, mb1, mw2, mb2,
                                  sw1, sb1, msw1, msb1, nid, w_sig, decay, sbias, mbias);
    for (int t = 0; t < TT; ++t) {
        k_step<<<dim3(NN / 64, BSZ), 128, 0, stream>>>(
            t, (t > 0) ? 1 : 0,
            msgb[t & 1], msgb[(t + 1) & 1],
            magb[(t + 1) & 1], magb[t & 1],
            h_g, conn, w_sig, sbias, mbias, decay, dbw, dgw,
            winT, sw2h, mwhh, mw2h, sb2, msb2, out_mem, heb);
    }
    k_fin<<<4096, 256, 0, stream>>>(heb, magb[(TT - 1) & 1], w_sig, conn, h_g, out_h, out_heb);
}
